// LIFConv2dLayer_39427799777436
// MI455X (gfx1250) — matrix operand order repaired and checked by reading, not compiled, not run on hardware
//
#include <hip/hip_runtime.h>

// ---------------------------------------------------------------------------
// LIF Conv2d timestep for MI455X (gfx1250, wave32, WMMA).
//   P_new = a*P + (1-a)*Q                  (bf16, channels-last, lives in L2)
//   U     = conv5x5(P_new, W) + bias - g*R (implicit GEMM, v_wmma_f32_16x16x32_bf16)
//   S     = U > 1
//   readout = 0.5*(maxpool2x2(S) @ w_sign^T) + 0.5
// input_t / BETA are dead (Q_new is not an output).
// Conv WG = 128 thr (4 waves) x (one (b,y) row, 64 x, 32 co): dual accumulators
// per wave -> 2 WMMAs per A-fragment load; A tile staged via async-to-LDS.
// ---------------------------------------------------------------------------

#define ALPHA_C 0.95122942f   // exp(-1e-3/20e-3)
#define OMA_C   0.04877058f   // 1 - ALPHA
#define GAMMA_C 0.70493454f   // exp(-1e-3/2.86e-3)

#define BB   64
#define CIN  64
#define HH   64
#define WW   64
#define COUT 128

#ifndef __has_builtin
#define __has_builtin(x) 0
#endif

#if __has_builtin(__builtin_amdgcn_global_load_async_to_lds_b128)
#define STAGE_ASYNC_B128 1
#endif

typedef __attribute__((ext_vector_type(16))) __bf16 v16bf;
typedef __attribute__((ext_vector_type(8)))  float  v8f;
typedef __attribute__((ext_vector_type(4)))  int    i32x4;

#define AS_GLOBAL(p) ((__attribute__((address_space(1))) i32x4*)(void*)(p))
#define AS_SHARED(p) ((__attribute__((address_space(3))) i32x4*)(void*)(p))

__device__ __forceinline__ unsigned short f2bf(float x) {
  union { float f; unsigned int u; } v; v.f = x;
  unsigned int r = v.u + 0x7FFFu + ((v.u >> 16) & 1u);  // round-nearest-even
  return (unsigned short)(r >> 16);
}

// ---------------------------------------------------------------------------
// Kernel 1: P_new = a*P + (1-a)*Q, f32 NCHW -> bf16 channels-last [b][y][x][ci]
// (LDS transpose so both global read and global write are coalesced)
// ---------------------------------------------------------------------------
__global__ __launch_bounds__(256) void trace_transpose_kernel(
    const float* __restrict__ P, const float* __restrict__ Q,
    unsigned short* __restrict__ pnewCL) {
  __shared__ unsigned short ldsT[CIN * 68];  // [ci][x], stride 68 (pad, 8B align)
  const int b = blockIdx.x >> 6;
  const int y = blockIdx.x & 63;
  const int t = threadIdx.x;
  const float4* __restrict__ P4 = (const float4*)P;
  const float4* __restrict__ Q4 = (const float4*)Q;
#pragma unroll
  for (int k = 0; k < 4; ++k) {
    int s  = t + (k << 8);       // 1024 float4 slots = 64ci x 16xq
    int ci = s >> 4;
    int xq = s & 15;
    int g  = ((b * CIN + ci) * HH + y) * 16 + xq;
    float4 p = P4[g];
    float4 q = Q4[g];
    float4 pn;
    pn.x = ALPHA_C * p.x + OMA_C * q.x;
    pn.y = ALPHA_C * p.y + OMA_C * q.y;
    pn.z = ALPHA_C * p.z + OMA_C * q.z;
    pn.w = ALPHA_C * p.w + OMA_C * q.w;
    uint2 u2;
    u2.x = (unsigned int)f2bf(pn.x) | ((unsigned int)f2bf(pn.y) << 16);
    u2.y = (unsigned int)f2bf(pn.z) | ((unsigned int)f2bf(pn.w) << 16);
    *(uint2*)(&ldsT[ci * 68 + (xq << 2)]) = u2;
  }
  __syncthreads();
  {
    int x  = t >> 2;
    int c0 = (t & 3) << 4;       // 16 channels per thread -> 32B store
    union { unsigned short s[16]; uint4 q[2]; } w;
#pragma unroll
    for (int j = 0; j < 16; ++j) w.s[j] = ldsT[(c0 + j) * 68 + x];
    int eb = (((b * HH + y) * WW + x) * CIN + c0);
    uint4* dst = (uint4*)(pnewCL + eb);
    dst[0] = w.q[0];
    dst[1] = w.q[1];
  }
}

// ---------------------------------------------------------------------------
// Kernel 2: weights f32 [Cout][Cin][5][5] -> bf16 in WMMA-B fragment order:
//   [coTile16(8)][tap(25)][kchunk(2)][lane(32)][j(16)]
// lane&15 = N (co), lane>>4 = h; lane holds K {8h..8h+7, 16+8h..16+8h+7} (ci).
// A B-fragment load is then 2 coalesced global_load_b128 per wave (L2-hot).
// ---------------------------------------------------------------------------
__global__ __launch_bounds__(256) void weight_frag_kernel(
    const float* __restrict__ Wt, unsigned short* __restrict__ wf) {
  int e = blockIdx.x * 256 + threadIdx.x;
  if (e >= 8 * 25 * 2 * 32 * 16) return;
  int j    = e & 15;
  int lane = (e >> 4) & 31;
  int c    = (e >> 9) & 1;
  int tap  = (e >> 10) % 25;
  int coT  = e / 25600;
  int co = (coT << 4) + (lane & 15);
  int ci = (c << 5) + ((j >> 3) << 4) + ((lane >> 4) << 3) + (j & 7);
  wf[e] = f2bf(Wt[(co * CIN + ci) * 25 + tap]);
}

// ---------------------------------------------------------------------------
// Kernel 3: implicit-GEMM conv + LIF epilogue.
// Workgroup = 128 thr (4 waves): one (b,y) row, 64 x positions, 32 channels.
// Wave w owns x tile [16w,16w+16) with TWO 16-co accumulators (A reuse x2):
// 100 k-steps of wmma_f32_16x16x32_bf16 per wave.
// A staged in LDS [ky(5)][col(68)][ci(64)] bf16 via GLOBAL_LOAD_ASYNC_TO_LDS;
// fragment load = 2x ds_load_b128 matching the documented 16-bit A layout.
// ---------------------------------------------------------------------------
__global__ __launch_bounds__(128) void lif_conv_wmma_kernel(
    const unsigned short* __restrict__ pnewCL,
    const unsigned short* __restrict__ wf,
    const float* __restrict__ R,
    const float* __restrict__ bias,
    float* __restrict__ Sout,
    float* __restrict__ Uout) {
  __shared__ unsigned short As[5 * 68 * CIN];  // 21760 elems = 43.5 KB

  const int coT   = blockIdx.x & 3;            // 32-channel tile
  const int rowId = blockIdx.x >> 2;
  const int y = rowId & 63;
  const int b = rowId >> 6;
  const int tid = threadIdx.x;

  // ---- Stage A tile: rows y-2..y+2, x in [-2,66), zero-padded halo ----
  {
    uint4* dstA = (uint4*)As;
    for (int i = tid; i < 2720; i += 128) {      // 2720 x 16B chunks
      int ci8 = i & 7;
      int col = (i >> 3) % 68;
      int r   = i / (68 * 8);
      int yy = y + r - 2;
      int xx = col - 2;
      bool inb = ((unsigned)yy < 64u) && ((unsigned)xx < 64u);
      const unsigned short* src =
          pnewCL + (((b * HH + yy) * WW + xx) * CIN + (ci8 << 3));
#if defined(STAGE_ASYNC_B128)
      if (inb) {
        __builtin_amdgcn_global_load_async_to_lds_b128(
            AS_GLOBAL(src), AS_SHARED(&dstA[i]), 0, 0);
      } else {
        uint4 z = {0u, 0u, 0u, 0u};
        dstA[i] = z;
      }
#else
      uint4 v = {0u, 0u, 0u, 0u};
      if (inb) v = *(const uint4*)src;
      dstA[i] = v;
#endif
    }
  }
#if defined(STAGE_ASYNC_B128)
  asm volatile("s_wait_asynccnt 0x0" ::: "memory");
#endif
  __syncthreads();

  const int lane = tid & 31;
  const int wave = tid >> 5;
  const int x0   = wave << 4;
  const int m    = lane & 15;
  const int half = lane >> 4;

  v8f acc0 = {0.f, 0.f, 0.f, 0.f, 0.f, 0.f, 0.f, 0.f};
  v8f acc1 = {0.f, 0.f, 0.f, 0.f, 0.f, 0.f, 0.f, 0.f};
  union FragU { uint4 q[2]; v16bf v; };

  const unsigned short* __restrict__ wfT0 = wf + (coT * 2) * 25600;
  const unsigned short* __restrict__ wfT1 = wfT0 + 25600;

#pragma unroll
  for (int ky = 0; ky < 5; ++ky) {
#pragma unroll
    for (int kx = 0; kx < 5; ++kx) {
      const int tap = ky * 5 + kx;
      const int colBase = ((ky * 68) + (x0 + m + kx)) << 6;  // element index
#pragma unroll
      for (int c = 0; c < 2; ++c) {
        FragU a, b0, b1;
        // A: lane<16 -> K {0..7,16..23}; lane>=16 -> K {8..15,24..31}
        const unsigned short* ap = As + colBase + (c << 5) + (half << 3);
        a.q[0] = *(const uint4*)(ap);        // K run 0
        a.q[1] = *(const uint4*)(ap + 16);   // K run 1 (+16 ci)
        // B: fragment-ordered weights, 32B contiguous per lane (WGP$/L2-hot)
        const int boff = ((((tap << 1) + c) << 5) + lane) * 16;
        b0.q[0] = *(const uint4*)(wfT0 + boff);
        b0.q[1] = *(const uint4*)(wfT0 + boff + 8);
        b1.q[0] = *(const uint4*)(wfT1 + boff);
        b1.q[1] = *(const uint4*)(wfT1 + boff + 8);
        acc0 = __builtin_amdgcn_wmma_f32_16x16x32_bf16(
            false, a.v, false, b0.v, (short)0, acc0, false, false);
        acc1 = __builtin_amdgcn_wmma_f32_16x16x32_bf16(
            false, a.v, false, b1.v, (short)0, acc1, false, false);
      }
    }
  }

  // ---- Epilogue: U = acc + bias - g*R ; S = U > 1 (for both 16-co tiles) ----
  // C layout: N = lane&15 (co), M = r + 8*half (x offset) -> 8 consecutive x.
  const int xb = x0 + (half << 3);
  auto epilogue = [&](v8f acc, int co) {
    const int g  = ((b * COUT + co) * HH + y) * WW + xb;
    const float bv = bias[co];
    float4 r0 = *(const float4*)(R + g);
    float4 r1 = *(const float4*)(R + g + 4);
    float u0 = acc[0] + bv - GAMMA_C * r0.x;
    float u1 = acc[1] + bv - GAMMA_C * r0.y;
    float u2 = acc[2] + bv - GAMMA_C * r0.z;
    float u3 = acc[3] + bv - GAMMA_C * r0.w;
    float u4 = acc[4] + bv - GAMMA_C * r1.x;
    float u5 = acc[5] + bv - GAMMA_C * r1.y;
    float u6 = acc[6] + bv - GAMMA_C * r1.z;
    float u7 = acc[7] + bv - GAMMA_C * r1.w;
    float4 uv0 = {u0, u1, u2, u3};
    float4 uv1 = {u4, u5, u6, u7};
    float4 sv0 = {u0 > 1.f ? 1.f : 0.f, u1 > 1.f ? 1.f : 0.f,
                  u2 > 1.f ? 1.f : 0.f, u3 > 1.f ? 1.f : 0.f};
    float4 sv1 = {u4 > 1.f ? 1.f : 0.f, u5 > 1.f ? 1.f : 0.f,
                  u6 > 1.f ? 1.f : 0.f, u7 > 1.f ? 1.f : 0.f};
    *(float4*)(Uout + g)     = uv0;
    *(float4*)(Uout + g + 4) = uv1;
    *(float4*)(Sout + g)     = sv0;
    *(float4*)(Sout + g + 4) = sv1;
  };
  const int co0 = (coT << 5) + m;
  epilogue(acc0, co0);
  epilogue(acc1, co0 + 16);
}

// ---------------------------------------------------------------------------
// Kernel 4: fused maxpool2x2 + readout partials. Each S element read once.
// Block = (b, chunk of 2048 pooled idx); deterministic block partials to ws.
// ---------------------------------------------------------------------------
__global__ __launch_bounds__(256) void pool_readout_kernel(
    const float* __restrict__ S, const float* __restrict__ wsgn,
    float* __restrict__ partial) {
  const int b     = blockIdx.x >> 6;
  const int chunk = blockIdx.x & 63;
  const int t = threadIdx.x;
  const int pbase = (chunk << 11) + (t << 3);   // 8 pooled per thread
  const int co  = pbase >> 10;
  const int py  = (pbase >> 5) & 31;
  const int px0 = pbase & 31;
  const float* row0 = S + ((b * COUT + co) * HH + (py << 1)) * WW + (px0 << 1);
  const float* row1 = row0 + WW;
  float pooled[8];
#pragma unroll
  for (int q = 0; q < 4; ++q) {
    float4 a = *(const float4*)(row0 + 4 * q);
    float4 c = *(const float4*)(row1 + 4 * q);
    pooled[2 * q]     = fmaxf(fmaxf(a.x, a.y), fmaxf(c.x, c.y));
    pooled[2 * q + 1] = fmaxf(fmaxf(a.z, a.w), fmaxf(c.z, c.w));
  }
  float accv[10];
#pragma unroll
  for (int o = 0; o < 10; ++o) {
    const float* wrow = wsgn + o * 131072 + pbase;
    float4 w0 = *(const float4*)(wrow);
    float4 w1 = *(const float4*)(wrow + 4);
    accv[o] = pooled[0] * w0.x + pooled[1] * w0.y + pooled[2] * w0.z +
              pooled[3] * w0.w + pooled[4] * w1.x + pooled[5] * w1.y +
              pooled[6] * w1.z + pooled[7] * w1.w;
  }
  __shared__ float red[256];
#pragma unroll 1
  for (int o = 0; o < 10; ++o) {
    red[t] = accv[o];
    __syncthreads();
    for (int s = 128; s > 0; s >>= 1) {
      if (t < s) red[t] += red[t + s];
      __syncthreads();
    }
    if (t == 0) partial[blockIdx.x * 10 + o] = red[0];
    __syncthreads();
  }
}

// Kernel 5: deterministic final reduction: readout = 0.5*sum + 0.5
__global__ void readout_final_kernel(const float* __restrict__ partial,
                                     float* __restrict__ ro) {
  int t = blockIdx.x * 64 + threadIdx.x;
  if (t < 640) {
    int b = t / 10, o = t % 10;
    float s = 0.f;
    for (int c = 0; c < 64; ++c) s += partial[(b * 64 + c) * 10 + o];
    ro[t] = 0.5f * s + 0.5f;
  }
}

// ---------------------------------------------------------------------------
extern "C" void kernel_launch(void* const* d_in, const int* in_sizes, int n_in,
                              void* d_out, int out_size, void* d_ws, size_t ws_size,
                              hipStream_t stream) {
  // inputs: 0=input_t(unused), 1=P, 2=Q, 3=R, 4=weights, 5=bias, 6=w_sign
  const float* P    = (const float*)d_in[1];
  const float* Q    = (const float*)d_in[2];
  const float* R    = (const float*)d_in[3];
  const float* Wt   = (const float*)d_in[4];
  const float* bias = (const float*)d_in[5];
  const float* wsgn = (const float*)d_in[6];

  float* Sout = (float*)d_out;                 // 33,554,432 floats
  float* Uout = Sout + 33554432;               // 33,554,432 floats
  float* ro   = Sout + 67108864;               // 640 floats

  unsigned short* pnewCL  = (unsigned short*)d_ws;        // 16,777,216 bf16
  unsigned short* wf      = pnewCL + 16777216;            // 204,800 bf16
  float*          partial = (float*)(wf + 204800);        // 40,960 f32

  trace_transpose_kernel<<<BB * HH, 256, 0, stream>>>(P, Q, pnewCL);
  weight_frag_kernel<<<800, 256, 0, stream>>>(Wt, wf);
  lif_conv_wmma_kernel<<<BB * HH * (COUT / 32), 128, 0, stream>>>(
      pnewCL, wf, R, bias, Sout, Uout);
  pool_readout_kernel<<<BB * 64, 256, 0, stream>>>(Sout, wsgn, partial);
  readout_final_kernel<<<10, 64, 0, stream>>>(partial, ro);
}
